// Transformer_35708358099230
// MI455X (gfx1250) — compile-verified, matrix operand order checked by reading, not run on hardware
//
#include <hip/hip_runtime.h>
#include <hip/hip_bf16.h>
#include <math.h>

// ---------------------------------------------------------------------------
// CDNA5 (gfx1250, wave32) transformer block.
// bf16 WMMA GEMMs, f32 accumulate, async global->LDS double-buffered staging.
// ---------------------------------------------------------------------------

typedef __attribute__((ext_vector_type(16))) __bf16 v16bf;
typedef __attribute__((ext_vector_type(8)))  __bf16 v8bf;
typedef __attribute__((ext_vector_type(8)))  float  v8f;
typedef int v4i __attribute__((vector_size(16)));

#define TB 256
static constexpr int BM = 128, BN = 256, BK = 32, LDP = 40; // pitch 40 bf16 = 80B rows (16B aligned)

__device__ __forceinline__ __bf16 f2bf(float f) {
  unsigned u = __float_as_uint(f);
  unsigned r = (u + 0x7FFFu + ((u >> 16) & 1u)) >> 16;
  unsigned short s = (unsigned short)r;
  return *reinterpret_cast<__bf16*>(&s);
}

// 16B global -> LDS async copy (ASYNCcnt path). Builtin if available, else ISA asm.
__device__ __forceinline__ void async_copy16(void* lds, const void* g) {
#if defined(__HIP_DEVICE_COMPILE__)
#if __has_builtin(__builtin_amdgcn_global_load_async_to_lds_b128)
  typedef __attribute__((address_space(1))) v4i* gp4;
  typedef __attribute__((address_space(3))) v4i* lp4;
  __builtin_amdgcn_global_load_async_to_lds_b128(
      (gp4)(uintptr_t)g,
      (lp4)(unsigned)(uintptr_t)lds, 0, 0);
#else
  unsigned lo = (unsigned)(uintptr_t)lds;  // generic LDS addr low 32 bits == LDS offset
  asm volatile("global_load_async_to_lds_b128 %0, %1, off"
               :: "v"(lo), "v"(g) : "memory");
#endif
#endif
}

__device__ __forceinline__ void wait_async0() {
#if defined(__HIP_DEVICE_COMPILE__)
#if __has_builtin(__builtin_amdgcn_s_wait_asynccnt)
  __builtin_amdgcn_s_wait_asynccnt(0);
#else
  asm volatile("s_wait_asynccnt 0" ::: "memory");
#endif
#endif
}

// ---------------------------------------------------------------------------
// Tiled bf16 WMMA GEMM: C = A[M][K] * B([N][K]) + bias + resid, opt ReLU.
// Block tile 128x256, 8 waves, wave tile 64x64 -> 16 wmma / K-step.
// Double-buffered LDS, async global->LDS staging.
// ---------------------------------------------------------------------------
template<bool RELU>
__global__ __launch_bounds__(TB)
void gemm_bf16_wmma(const __bf16* __restrict__ A, int lda, long sa_b, long sa_h,
                    const __bf16* __restrict__ Bm, int ldb, long sb_b, long sb_h,
                    float* __restrict__ outF, __bf16* __restrict__ outB,
                    int ldc, long sc_b, long sc_h,
                    const float* __restrict__ bias,
                    const float* __restrict__ resid, int ldr,
                    int K, int batch_h)
{
  __shared__ __bf16 sA[2][BM * LDP];
  __shared__ __bf16 sB[2][BN * LDP];

  const int tid  = threadIdx.x;
  const int lane = tid & 31;
  const int wave = tid >> 5;
  const int wm   = wave >> 2;      // 0..1 : 64-row slab
  const int wn   = wave & 3;       // 0..3 : 64-col slab
  const int half = lane >> 4;      // ISA frag layout: upper/lower 16 lanes
  const int l16  = lane & 15;

  const int zb = blockIdx.z / batch_h;
  const int zh = blockIdx.z % batch_h;
  A  += (long)zb * sa_b + (long)zh * sa_h;
  Bm += (long)zb * sb_b + (long)zh * sb_h;
  const long coff = (long)zb * sc_b + (long)zh * sc_h;

  const int m0 = blockIdx.y * BM;
  const int n0 = blockIdx.x * BN;

  v8f acc[4][4] = {};

  // issue async loads of one K-tile into buffer `buf`
  auto stage = [&](int buf, int kt) {
#pragma unroll
    for (int s = 0; s < 2; ++s) {            // A: 128x32 = 512 16B chunks
      int c = tid + TB * s;
      int row = c >> 2, col8 = (c & 3) << 3;
      async_copy16(&sA[buf][row * LDP + col8],
                   A + (long)(m0 + row) * lda + kt + col8);
    }
#pragma unroll
    for (int s = 0; s < 4; ++s) {            // B: 256x32 = 1024 16B chunks
      int c = tid + TB * s;
      int row = c >> 2, col8 = (c & 3) << 3;
      async_copy16(&sB[buf][row * LDP + col8],
                   Bm + (long)(n0 + row) * ldb + kt + col8);
    }
  };

  stage(0, 0);
  int buf = 0;

  for (int kt = 0; kt < K; kt += BK) {
    wait_async0();        // this wave's async tile complete
    __syncthreads();      // whole WG's tile complete / prior readers done
    if (kt + BK < K) stage(buf ^ 1, kt + BK);   // prefetch next tile async

    const int koff = half * 8;
    v16bf af[4], bfr[4];
#pragma unroll
    for (int i = 0; i < 4; ++i) {
      const __bf16* p = &sA[buf][(wm * 64 + i * 16 + l16) * LDP + koff];
      v8bf lo = *reinterpret_cast<const v8bf*>(p);
      v8bf hi = *reinterpret_cast<const v8bf*>(p + 16);
      af[i] = __builtin_shufflevector(lo, hi, 0,1,2,3,4,5,6,7,8,9,10,11,12,13,14,15);
    }
#pragma unroll
    for (int j = 0; j < 4; ++j) {
      const __bf16* p = &sB[buf][(wn * 64 + j * 16 + l16) * LDP + koff];
      v8bf lo = *reinterpret_cast<const v8bf*>(p);
      v8bf hi = *reinterpret_cast<const v8bf*>(p + 16);
      bfr[j] = __builtin_shufflevector(lo, hi, 0,1,2,3,4,5,6,7,8,9,10,11,12,13,14,15);
    }
#pragma unroll
    for (int i = 0; i < 4; ++i)
#pragma unroll
      for (int j = 0; j < 4; ++j)
        acc[i][j] = __builtin_amdgcn_wmma_f32_16x16x32_bf16(
            false, af[i], false, bfr[j], (short)0, acc[i][j], false, false);

    buf ^= 1;
  }

  // ---- epilogue: bias / residual / ReLU; f32 and/or bf16 stores ----
#pragma unroll
  for (int i = 0; i < 4; ++i) {
#pragma unroll
    for (int j = 0; j < 4; ++j) {
      const int colg = n0 + wn * 64 + j * 16 + l16;
      const float bsv = bias ? bias[colg] : 0.0f;
#pragma unroll
      for (int e = 0; e < 8; ++e) {
        const int rowg = m0 + wm * 64 + i * 16 + e + 8 * half;
        float v = acc[i][j][e] + bsv;
        if (resid) v += resid[(long)rowg * ldr + colg];
        if (RELU)  v = fmaxf(v, 0.0f);
        const long ci = coff + (long)rowg * ldc + colg;
        if (outF) outF[ci] = v;
        if (outB) outB[ci] = f2bf(v);
      }
    }
  }
}

// ---------------------------------------------------------------------------
// LayerNorm row kernel (f32 in -> bf16 out), one 256-thread WG per row.
// ---------------------------------------------------------------------------
__global__ __launch_bounds__(TB)
void ln_bf16_kernel(const float* __restrict__ x, const float* __restrict__ g,
                    const float* __restrict__ bta, __bf16* __restrict__ y, int D)
{
  __shared__ float s1[TB], s2[TB];
  const long row = blockIdx.x;
  const float* xr = x + row * D;
  float sum = 0.f, sq = 0.f;
  for (int c = threadIdx.x; c < D; c += TB) { float v = xr[c]; sum += v; sq += v * v; }
  s1[threadIdx.x] = sum; s2[threadIdx.x] = sq;
  __syncthreads();
  for (int off = TB / 2; off > 0; off >>= 1) {
    if (threadIdx.x < off) {
      s1[threadIdx.x] += s1[threadIdx.x + off];
      s2[threadIdx.x] += s2[threadIdx.x + off];
    }
    __syncthreads();
  }
  const float mean = s1[0] / D;
  const float var  = s2[0] / D - mean * mean;
  const float rs   = rsqrtf(var + 1e-5f);
  for (int c = threadIdx.x; c < D; c += TB)
    y[row * D + c] = f2bf((xr[c] - mean) * rs * g[c] + bta[c]);
}

// ---------------------------------------------------------------------------
// Row softmax with scale (f32 in -> bf16 out), one WG per row of length L.
// ---------------------------------------------------------------------------
__global__ __launch_bounds__(TB)
void softmax_bf16_kernel(const float* __restrict__ s, __bf16* __restrict__ w,
                         int L, float scale)
{
  __shared__ float red[TB];
  const long row = blockIdx.x;
  const float* sr = s + row * (long)L;
  float m = -__builtin_inff();
  for (int c = threadIdx.x; c < L; c += TB) m = fmaxf(m, sr[c]);
  red[threadIdx.x] = m;
  __syncthreads();
  for (int off = TB / 2; off > 0; off >>= 1) {
    if (threadIdx.x < off) red[threadIdx.x] = fmaxf(red[threadIdx.x], red[threadIdx.x + off]);
    __syncthreads();
  }
  m = red[0] * scale;
  __syncthreads();
  float sum = 0.f;
  for (int c = threadIdx.x; c < L; c += TB) sum += __expf(sr[c] * scale - m);
  red[threadIdx.x] = sum;
  __syncthreads();
  for (int off = TB / 2; off > 0; off >>= 1) {
    if (threadIdx.x < off) red[threadIdx.x] += red[threadIdx.x + off];
    __syncthreads();
  }
  const float inv = 1.0f / red[0];
  for (int c = threadIdx.x; c < L; c += TB)
    w[row * (long)L + c] = f2bf(__expf(sr[c] * scale - m) * inv);
}

// ---------------------------------------------------------------------------
// Transpose + convert: f32 [R][C] -> bf16 [C][R] (weights).
// ---------------------------------------------------------------------------
__global__ __launch_bounds__(TB)
void transpose_f32_to_bf16(const float* __restrict__ in, __bf16* __restrict__ out,
                           int R, int C)
{
  __shared__ float t[32][33];
  const int c0 = blockIdx.x * 32, r0 = blockIdx.y * 32;
  const int tx = threadIdx.x, ty = threadIdx.y;   // block (32,8)
#pragma unroll
  for (int i = 0; i < 32; i += 8)
    t[ty + i][tx] = in[(long)(r0 + ty + i) * C + c0 + tx];
  __syncthreads();
#pragma unroll
  for (int i = 0; i < 32; i += 8)
    out[(long)(c0 + ty + i) * R + r0 + tx] = f2bf(t[tx][ty + i]);
}

// ---------------------------------------------------------------------------
// Batched bf16 transpose: per z=(b,h), in [R][C] (row stride ldin) -> out [C][R].
// Used to pre-transpose V so the att GEMM consumes B in [N][K] form.
// ---------------------------------------------------------------------------
__global__ __launch_bounds__(TB)
void transpose_bf16_batched(const __bf16* __restrict__ in, __bf16* __restrict__ out,
                            int ldin, int ldout, long sin_b, long sin_h,
                            long sout, int batch_h)
{
  __shared__ __bf16 t[32][33];
  const int zb = blockIdx.z / batch_h, zh = blockIdx.z % batch_h;
  in  += (long)zb * sin_b + (long)zh * sin_h;
  out += (long)blockIdx.z * sout;
  const int c0 = blockIdx.x * 32, r0 = blockIdx.y * 32;
  const int tx = threadIdx.x, ty = threadIdx.y;   // block (32,8)
#pragma unroll
  for (int i = 0; i < 32; i += 8)
    t[ty + i][tx] = in[(long)(r0 + ty + i) * ldin + c0 + tx];
  __syncthreads();
#pragma unroll
  for (int i = 0; i < 32; i += 8)
    out[(long)(c0 + ty + i) * ldout + r0 + tx] = t[tx][ty + i];
}

// ---------------------------------------------------------------------------
extern "C" void kernel_launch(void* const* d_in, const int* in_sizes, int n_in,
                              void* d_out, int out_size, void* d_ws, size_t ws_size,
                              hipStream_t stream)
{
  (void)in_sizes; (void)n_in; (void)out_size; (void)ws_size;
  constexpr int Bb = 4, Nn = 1024, D = 768, H = 12;
  constexpr int DH = D * H;            // 9216
  constexpr int M  = Bb * Nn;          // 4096
  constexpr int D4 = 4 * D;            // 3072

  const float* x    = (const float*)d_in[0];
  const float* Wq   = (const float*)d_in[1];
  const float* Wk   = (const float*)d_in[2];
  const float* Wv   = (const float*)d_in[3];
  const float* Wu   = (const float*)d_in[4];
  const float* bu   = (const float*)d_in[5];
  const float* ln1g = (const float*)d_in[6];
  const float* ln1b = (const float*)d_in[7];
  const float* ln2g = (const float*)d_in[8];
  const float* ln2b = (const float*)d_in[9];
  const float* W1   = (const float*)d_in[10];
  const float* b1   = (const float*)d_in[11];
  const float* W2   = (const float*)d_in[12];
  const float* b2   = (const float*)d_in[13];
  float* out = (float*)d_out;

  // --- workspace carve-out ---
  char* base = (char*)d_ws;
  size_t off = 0;
  auto alloc = [&](size_t bytes) -> void* {
    void* p = base + off;
    off += (bytes + 255) & ~(size_t)255;
    return p;
  };
  __bf16* yln  = (__bf16*)alloc((size_t)M * D * 2);
  __bf16* WqT  = (__bf16*)alloc((size_t)DH * D * 2);
  __bf16* WkT  = (__bf16*)alloc((size_t)DH * D * 2);
  __bf16* WvT  = (__bf16*)alloc((size_t)DH * D * 2);
  __bf16* WuT  = (__bf16*)alloc((size_t)D * DH * 2);
  __bf16* W1T  = (__bf16*)alloc((size_t)D4 * D * 2);
  __bf16* W2T  = (__bf16*)alloc((size_t)D * D4 * 2);
  __bf16* Qb   = (__bf16*)alloc((size_t)M * DH * 2);
  __bf16* Kb   = (__bf16*)alloc((size_t)M * DH * 2);
  __bf16* Vb   = (__bf16*)alloc((size_t)M * DH * 2);
  __bf16* VT   = (__bf16*)alloc((size_t)Bb * H * D * Nn * 2);
  float*  sc   = (float*) alloc((size_t)Bb * H * Nn * Nn * 4);
  __bf16* wsm  = (__bf16*)alloc((size_t)Bb * H * Nn * Nn * 2);
  __bf16* attb = (__bf16*)alloc((size_t)M * DH * 2);
  float*  x1   = (float*) alloc((size_t)M * D * 4);
  __bf16* y2   = (__bf16*)alloc((size_t)M * D * 2);
  __bf16* h1   = (__bf16*)alloc((size_t)M * D4 * 2);

  const dim3 tb(TB), tt(32, 8);

  // 1) weight prep: f32 [K][N] -> bf16 [N][K]
  auto T = [&](const float* in, __bf16* o, int R, int C) {
    transpose_f32_to_bf16<<<dim3(C / 32, R / 32), tt, 0, stream>>>(in, o, R, C);
  };
  T(Wq, WqT, D, DH);  T(Wk, WkT, D, DH);  T(Wv, WvT, D, DH);
  T(Wu, WuT, DH, D);  T(W1, W1T, D, D4);  T(W2, W2T, D4, D);

  // 2) LN1 -> bf16 activations
  ln_bf16_kernel<<<M, tb, 0, stream>>>(x, ln1g, ln1b, yln, D);

  // 3) QKV projections: [4096x768] x [768x9216] -> bf16
  auto qkv = [&](const __bf16* Wt, __bf16* o) {
    gemm_bf16_wmma<false><<<dim3(DH / BN, M / BM, 1), tb, 0, stream>>>(
        yln, D, 0, 0, Wt, D, 0, 0,
        nullptr, o, DH, 0, 0, nullptr, nullptr, 0, D, 1);
  };
  qkv(WqT, Qb);  qkv(WkT, Kb);  qkv(WvT, Vb);

  // 3b) pre-transpose V per (b,h): [j][d] -> [d][j] so att-GEMM B is [N][K]
  transpose_bf16_batched<<<dim3(D / 32, Nn / 32, Bb * H), tt, 0, stream>>>(
      Vb, VT, DH, Nn, (long)Nn * DH, D, (long)D * Nn, H);

  // 4) scores = Q K^T per (b,h): 48 x (1024x1024x768); B=[N][K] = K rows
  gemm_bf16_wmma<false><<<dim3(Nn / BN, Nn / BM, Bb * H), tb, 0, stream>>>(
      Qb, DH, (long)Nn * DH, D,
      Kb, DH, (long)Nn * DH, D,
      sc, nullptr, Nn, (long)H * Nn * Nn, (long)Nn * Nn,
      nullptr, nullptr, 0, D, H);

  // 5) softmax rows (scale 1/sqrt(768)) -> bf16 probabilities
  softmax_bf16_kernel<<<Bb * H * Nn, tb, 0, stream>>>(sc, wsm, Nn, rsqrtf((float)D));

  // 6) att = w V per (b,h): 48 x (1024x768x1024), B = VT [d][j]
  gemm_bf16_wmma<false><<<dim3(D / BN, Nn / BM, Bb * H), tb, 0, stream>>>(
      wsm, Nn, (long)H * Nn * Nn, (long)Nn * Nn,
      VT, Nn, (long)H * D * Nn, (long)D * Nn,
      nullptr, attb, DH, (long)Nn * DH, D,
      nullptr, nullptr, 0, Nn, H);

  // 7) out proj + bias + residual(x): [4096x9216] x [9216x768] -> f32 x1
  gemm_bf16_wmma<false><<<dim3(D / BN, M / BM, 1), tb, 0, stream>>>(
      attb, DH, 0, 0, WuT, DH, 0, 0,
      x1, nullptr, D, 0, 0, bu, x, D, DH, 1);

  // 8) LN2
  ln_bf16_kernel<<<M, tb, 0, stream>>>(x1, ln2g, ln2b, y2, D);

  // 9) MLP1 + bias + ReLU: [4096x768] x [768x3072] -> bf16 h1
  gemm_bf16_wmma<true><<<dim3(D4 / BN, M / BM, 1), tb, 0, stream>>>(
      y2, D, 0, 0, W1T, D, 0, 0,
      nullptr, h1, D4, 0, 0, b1, nullptr, 0, D, 1);

  // 10) MLP2 + bias + residual(x1) -> f32 output
  gemm_bf16_wmma<false><<<dim3(D / BN, M / BM, 1), tb, 0, stream>>>(
      h1, D4, 0, 0, W2T, D4, 0, 0,
      out, nullptr, D, 0, 0, b2, x1, D, D4, 1);
}
